// Attention3D_55843164782982
// MI455X (gfx1250) — matrix operand order repaired and checked by reading, not compiled, not run on hardware
//
#include <hip/hip_runtime.h>

#define N_TOK 4096
#define CDIM  128
#define HEADS 8
#define DHEAD 64
#define HID   512

typedef __attribute__((ext_vector_type(16))) __bf16 v16bf;
typedef __attribute__((ext_vector_type(8)))  float  v8f;

union V16 {
  v16bf v;
  uint4 q[2];
  unsigned short u[16];
};

union U8 {               // 8 bf16 packed for one b128 store
  uint4 q;
  unsigned short u[8];
};

static __device__ __forceinline__ unsigned short f2bfu(float f) {
  __bf16 h = (__bf16)f;                       // native v_cvt_pk_bf16_f32 on gfx1250
  return __builtin_bit_cast(unsigned short, h);
}

// ---------------------------------------------------------------------------
// Kernel 0: one-time operand prep (bf16 weights + transposed bf16 x)
//   Wq  [1536][128]  <- w_qkv
//   Wo  [128][512]   <- w_out
//   Xt  [4096][128]  <- x (transposed token-major)
// ---------------------------------------------------------------------------
__global__ __launch_bounds__(256) void prep(
    const float* __restrict__ x, const float* __restrict__ w_qkv,
    const float* __restrict__ w_out, unsigned short* __restrict__ Wq,
    unsigned short* __restrict__ Wo, unsigned short* __restrict__ Xt) {
  const int id = blockIdx.x * 256 + threadIdx.x;   // 786432 total
  if (id < 1536 * CDIM) {
    Wq[id] = f2bfu(w_qkv[id]);
  } else if (id < 1536 * CDIM + CDIM * HID) {
    const int e = id - 1536 * CDIM;
    Wo[e] = f2bfu(w_out[e]);
  } else {
    const int e = id - (1536 * CDIM + CDIM * HID); // Xt flat index
    const int c = e & (CDIM - 1);
    const int tok = e >> 7;
    Xt[e] = f2bfu(x[(size_t)c * N_TOK + tok]);
  }
}

// ---------------------------------------------------------------------------
// Kernel 1: qkv = Wq @ Xt^T  (M=1536, K=128, N=4096), scatter to Q/K/V (bf16)
// Q: [h][n][d] (*SCALE), K: [h][n][d], V: [h][d][n]
// Region (q/k/v) is uniform per m_tile: 0..31 -> Q, 32..63 -> K, 64..95 -> V.
// ---------------------------------------------------------------------------
__global__ __launch_bounds__(256) void qkv_proj(
    const unsigned short* __restrict__ Wq, const unsigned short* __restrict__ Xt,
    unsigned short* __restrict__ Qb, unsigned short* __restrict__ Kb,
    unsigned short* __restrict__ Vb) {
  const int lane   = threadIdx.x & 31;
  const int wave   = threadIdx.x >> 5;
  const int tile   = blockIdx.x * 8 + wave;   // 96*256 tiles total
  const int m_tile = tile >> 8;               // 0..95
  const int n_tile = tile & 255;              // 0..255
  const int mrow   = lane & 15;
  const int kbaseA = (lane >> 4) * 16;
  const int ncol   = lane & 15;
  const int kbaseB = (lane >> 4) * 16;
  const int token  = n_tile * 16 + ncol;

  v8f acc = {};
  const unsigned short* wrow = Wq + (size_t)(m_tile * 16 + mrow) * CDIM;
  const unsigned short* xrow = Xt + (size_t)token * CDIM;
#pragma unroll
  for (int ks = 0; ks < 4; ++ks) {
    V16 a;
    a.q[0] = *(const uint4*)(wrow + ks * 32 + kbaseA);
    a.q[1] = *(const uint4*)(wrow + ks * 32 + kbaseA + 8);
    V16 b;
    b.q[0] = *(const uint4*)(xrow + ks * 32 + kbaseB);
    b.q[1] = *(const uint4*)(xrow + ks * 32 + kbaseB + 8);
    acc = __builtin_amdgcn_wmma_f32_16x16x32_bf16(false, a.v, false, b.v,
                                                  (short)0, acc, false, false);
  }

  const int obase = m_tile * 16 + (lane >> 4) * 8;  // 8-aligned row group
  if (m_tile < 32) {                 // Q: one packed b128 store, pre-scaled
    const int h = obase >> 6, dbase = obase & 63;
    U8 st;
#pragma unroll
    for (int r = 0; r < 8; ++r) st.u[r] = f2bfu(acc[r] * 0.125f); // SCALE
    *(uint4*)(Qb + ((size_t)h * N_TOK + token) * DHEAD + dbase) = st.q;
  } else if (m_tile < 64) {          // K: one packed b128 store
    const int o2 = obase - HID, h = o2 >> 6, dbase = o2 & 63;
    U8 st;
#pragma unroll
    for (int r = 0; r < 8; ++r) st.u[r] = f2bfu(acc[r]);
    *(uint4*)(Kb + ((size_t)h * N_TOK + token) * DHEAD + dbase) = st.q;
  } else {                           // V: transposed layout, element stores
    const int o2 = obase - 2 * HID, h = o2 >> 6, dbase = o2 & 63;
#pragma unroll
    for (int r = 0; r < 8; ++r)
      Vb[((size_t)h * DHEAD + dbase + r) * N_TOK + token] = f2bfu(acc[r]);
  }
}

// ---------------------------------------------------------------------------
// Kernel 2: flash attention. 8 waves/block, 16 query rows per wave,
// K/V j-tiles (32 wide) staged in LDS via global_load_async_to_lds_b128,
// online softmax, O via WMMA. Output Ob: [n][512] bf16 (token-major hidden).
// ---------------------------------------------------------------------------
__global__ __launch_bounds__(256) void flash_attn(
    const unsigned short* __restrict__ Qb, const unsigned short* __restrict__ Kb,
    const unsigned short* __restrict__ Vb, unsigned short* __restrict__ Ob) {
  __shared__ alignas(16) unsigned short kt[32 * 64];     // [j][d]
  __shared__ alignas(16) unsigned short vt[64 * 32];     // [d][j]
  __shared__ alignas(16) unsigned short pbuf[8][16 * 32];// per-wave P transpose

  const int tid  = threadIdx.x;
  const int lane = tid & 31;
  const int wave = tid >> 5;
  const int h    = blockIdx.x >> 5;   // 8 heads
  const int rb   = blockIdx.x & 31;   // 32 row-blocks of 128
  const int i0   = rb * 128 + wave * 16;

  const int mrow   = lane & 15;
  const int kbaseA = (lane >> 4) * 16;
  const int ncol   = lane & 15;
  const int kbaseB = (lane >> 4) * 16;

  // Q tile (A layout, 16 x 64) — contiguous 16B chunks
  V16 aq0, aq1;
  {
    const unsigned short* qrow = Qb + ((size_t)h * N_TOK + (i0 + mrow)) * DHEAD;
    aq0.q[0] = *(const uint4*)(qrow + kbaseA);
    aq0.q[1] = *(const uint4*)(qrow + kbaseA + 8);
    aq1.q[0] = *(const uint4*)(qrow + 32 + kbaseA);
    aq1.q[1] = *(const uint4*)(qrow + 40 + kbaseA);
  }

  v8f o0 = {}, o1 = {}, o2 = {}, o3 = {};
  float m_run[8], l_run[8];
#pragma unroll
  for (int r = 0; r < 8; ++r) { m_run[r] = -1e30f; l_run[r] = 0.f; }

  const unsigned short* Kh = Kb + (size_t)h * N_TOK * DHEAD;
  const unsigned short* Vh = Vb + (size_t)h * DHEAD * N_TOK;

  // per-thread staging addresses (constant across the j loop)
  const int kidx = tid * 8;                 // K tile: 256 x 16B
  const int vrow = tid >> 2;                // V tile: 64 rows x 32 cols
  const int vcol = (tid & 3) * 8;
  const unsigned kdst = (unsigned)(size_t)(kt + kidx);            // LDS offset
  const unsigned vdst = (unsigned)(size_t)(vt + vrow * 32 + vcol);// LDS offset

  for (int j0 = 0; j0 < N_TOK; j0 += 32) {
    __syncthreads();
    {  // CDNA5 async copy: global -> LDS, tracked by ASYNCcnt
      const unsigned short* ksrc = Kh + (size_t)j0 * DHEAD + kidx;
      const unsigned short* vsrc = Vh + (size_t)vrow * N_TOK + j0 + vcol;
      asm volatile("global_load_async_to_lds_b128 %0, %1, off"
                   :: "v"(kdst), "v"(ksrc) : "memory");
      asm volatile("global_load_async_to_lds_b128 %0, %1, off"
                   :: "v"(vdst), "v"(vsrc) : "memory");
      if (j0 + 32 < N_TOK) {  // gfx1250 global_prefetch_b8 of next tiles
        __builtin_prefetch(Kh + (size_t)(j0 + 32) * DHEAD + kidx, 0, 1);
        __builtin_prefetch(Vh + (size_t)vrow * N_TOK + j0 + 32 + vcol, 0, 1);
      }
      asm volatile("s_wait_asynccnt 0" ::: "memory");
    }
    __syncthreads();

    // S = Q @ K^T for two 16-wide j sub-tiles (4 WMMAs)
    v8f s0 = {}, s1 = {};
    {
      V16 bk;
      bk.q[0] = *(const uint4*)(kt + ncol * 64 + kbaseB);
      bk.q[1] = *(const uint4*)(kt + ncol * 64 + kbaseB + 8);
      s0 = __builtin_amdgcn_wmma_f32_16x16x32_bf16(false, aq0.v, false, bk.v,
                                                   (short)0, s0, false, false);
      bk.q[0] = *(const uint4*)(kt + ncol * 64 + 32 + kbaseB);
      bk.q[1] = *(const uint4*)(kt + ncol * 64 + 32 + kbaseB + 8);
      s0 = __builtin_amdgcn_wmma_f32_16x16x32_bf16(false, aq1.v, false, bk.v,
                                                   (short)0, s0, false, false);
      bk.q[0] = *(const uint4*)(kt + (16 + ncol) * 64 + kbaseB);
      bk.q[1] = *(const uint4*)(kt + (16 + ncol) * 64 + kbaseB + 8);
      s1 = __builtin_amdgcn_wmma_f32_16x16x32_bf16(false, aq0.v, false, bk.v,
                                                   (short)0, s1, false, false);
      bk.q[0] = *(const uint4*)(kt + (16 + ncol) * 64 + 32 + kbaseB);
      bk.q[1] = *(const uint4*)(kt + (16 + ncol) * 64 + 32 + kbaseB + 8);
      s1 = __builtin_amdgcn_wmma_f32_16x16x32_bf16(false, aq1.v, false, bk.v,
                                                   (short)0, s1, false, false);
    }

    // online softmax (rows split across lane halves; reduce within 16 lanes)
    float scale[8], p0[8], p1[8];
#pragma unroll
    for (int r = 0; r < 8; ++r) {
      float t = fmaxf(s0[r], s1[r]);
#pragma unroll
      for (int mk = 1; mk <= 8; mk <<= 1) t = fmaxf(t, __shfl_xor(t, mk, 32));
      const float mnew = fmaxf(m_run[r], t);
      scale[r] = __expf(m_run[r] - mnew);
      p0[r] = __expf(s0[r] - mnew);
      p1[r] = __expf(s1[r] - mnew);
      float rs = p0[r] + p1[r];
#pragma unroll
      for (int mk = 1; mk <= 8; mk <<= 1) rs += __shfl_xor(rs, mk, 32);
      l_run[r] = l_run[r] * scale[r] + rs;
      m_run[r] = mnew;
    }
#pragma unroll
    for (int r = 0; r < 8; ++r) {
      o0[r] *= scale[r]; o1[r] *= scale[r];
      o2[r] *= scale[r]; o3[r] *= scale[r];
    }

    // P: C-layout -> A-layout via per-wave LDS round-trip
    unsigned short* pw = pbuf[wave];
#pragma unroll
    for (int r = 0; r < 8; ++r) {
      const int row = (lane >> 4) * 8 + r;
      pw[row * 32 + ncol]      = f2bfu(p0[r]);
      pw[row * 32 + 16 + ncol] = f2bfu(p1[r]);
    }
    V16 pa;
    pa.q[0] = *(const uint4*)(pw + mrow * 32 + kbaseA);
    pa.q[1] = *(const uint4*)(pw + mrow * 32 + kbaseA + 8);

    // O += P @ V  (4 WMMAs over d tiles)
    V16 bv;
    bv.q[0] = *(const uint4*)(vt + ncol * 32 + kbaseB);
    bv.q[1] = *(const uint4*)(vt + ncol * 32 + kbaseB + 8);
    o0 = __builtin_amdgcn_wmma_f32_16x16x32_bf16(false, pa.v, false, bv.v,
                                                 (short)0, o0, false, false);
    bv.q[0] = *(const uint4*)(vt + (16 + ncol) * 32 + kbaseB);
    bv.q[1] = *(const uint4*)(vt + (16 + ncol) * 32 + kbaseB + 8);
    o1 = __builtin_amdgcn_wmma_f32_16x16x32_bf16(false, pa.v, false, bv.v,
                                                 (short)0, o1, false, false);
    bv.q[0] = *(const uint4*)(vt + (32 + ncol) * 32 + kbaseB);
    bv.q[1] = *(const uint4*)(vt + (32 + ncol) * 32 + kbaseB + 8);
    o2 = __builtin_amdgcn_wmma_f32_16x16x32_bf16(false, pa.v, false, bv.v,
                                                 (short)0, o2, false, false);
    bv.q[0] = *(const uint4*)(vt + (48 + ncol) * 32 + kbaseB);
    bv.q[1] = *(const uint4*)(vt + (48 + ncol) * 32 + kbaseB + 8);
    o3 = __builtin_amdgcn_wmma_f32_16x16x32_bf16(false, pa.v, false, bv.v,
                                                 (short)0, o3, false, false);
  }

  // normalize and store token-major [n][512]
#pragma unroll
  for (int r = 0; r < 8; ++r) {
    const float inv = 1.0f / l_run[r];
    const int row = (lane >> 4) * 8 + r;
    const size_t base = (size_t)(i0 + row) * HID + h * DHEAD;
    Ob[base + 0  + ncol] = f2bfu(o0[r] * inv);
    Ob[base + 16 + ncol] = f2bfu(o1[r] * inv);
    Ob[base + 32 + ncol] = f2bfu(o2[r] * inv);
    Ob[base + 48 + ncol] = f2bfu(o3[r] * inv);
  }
}

// ---------------------------------------------------------------------------
// Kernel 3: out = Wo @ attn_out + b  (M=128, K=512, N=4096), f32 output
// ---------------------------------------------------------------------------
__global__ __launch_bounds__(256) void out_proj(
    const unsigned short* __restrict__ Ob, const unsigned short* __restrict__ Wo,
    const float* __restrict__ b_out, float* __restrict__ out) {
  const int lane   = threadIdx.x & 31;
  const int wave   = threadIdx.x >> 5;
  const int tile   = blockIdx.x * 8 + wave;   // 8*256 tiles
  const int m_tile = tile >> 8;               // 0..7
  const int n_tile = tile & 255;
  const int mrow   = lane & 15;
  const int kbaseA = (lane >> 4) * 16;
  const int ncol   = lane & 15;
  const int kbaseB = (lane >> 4) * 16;
  const int token  = n_tile * 16 + ncol;

  v8f acc = {};
  const unsigned short* wrow = Wo + (size_t)(m_tile * 16 + mrow) * HID;
  const unsigned short* orow = Ob + (size_t)token * HID;
#pragma unroll
  for (int ks = 0; ks < 16; ++ks) {
    V16 a;
    a.q[0] = *(const uint4*)(wrow + ks * 32 + kbaseA);
    a.q[1] = *(const uint4*)(wrow + ks * 32 + kbaseA + 8);
    V16 b;
    b.q[0] = *(const uint4*)(orow + ks * 32 + kbaseB);
    b.q[1] = *(const uint4*)(orow + ks * 32 + kbaseB + 8);
    acc = __builtin_amdgcn_wmma_f32_16x16x32_bf16(false, a.v, false, b.v,
                                                  (short)0, acc, false, false);
  }
#pragma unroll
  for (int r = 0; r < 8; ++r) {
    const int o = m_tile * 16 + (lane >> 4) * 8 + r;
    out[(size_t)o * N_TOK + token] = acc[r] + b_out[o];
  }
}

// ---------------------------------------------------------------------------
extern "C" void kernel_launch(void* const* d_in, const int* in_sizes, int n_in,
                              void* d_out, int out_size, void* d_ws, size_t ws_size,
                              hipStream_t stream) {
  const float* x     = (const float*)d_in[0];
  const float* w_qkv = (const float*)d_in[1];
  const float* w_out = (const float*)d_in[2];
  const float* b_out = (const float*)d_in[3];

  unsigned short* Qb = (unsigned short*)d_ws;                       // 4 MB
  unsigned short* Kb = Qb + (size_t)HEADS * N_TOK * DHEAD;          // 4 MB
  unsigned short* Vb = Kb + (size_t)HEADS * N_TOK * DHEAD;          // 4 MB
  unsigned short* Ob = Vb + (size_t)HEADS * N_TOK * DHEAD;          // 4 MB
  unsigned short* Wq = Ob + (size_t)N_TOK * HID;                    // 384 KB
  unsigned short* Wo = Wq + (size_t)1536 * CDIM;                    // 128 KB
  unsigned short* Xt = Wo + (size_t)CDIM * HID;                     // 1 MB

  prep<<<3072, 256, 0, stream>>>(x, w_qkv, w_out, Wq, Wo, Xt);
  qkv_proj<<<3072, 256, 0, stream>>>(Wq, Xt, Qb, Kb, Vb);
  flash_attn<<<256, 256, 0, stream>>>(Qb, Kb, Vb, Ob);
  out_proj<<<256, 256, 0, stream>>>(Ob, Wo, b_out, (float*)d_out);
}
